// SoftForwardWarp_78537771975202
// MI455X (gfx1250) — matrix operand order repaired and checked by reading, not compiled, not run on hardware
//
#include <hip/hip_runtime.h>
#include <stdint.h>

// SoftForwardWarp on MI455X (gfx1250).
// Exact factorization of the separable L1 softmax:
//   out[b,c,gx,gy] = sum_j U[gx,j] * V[gy,j] * x[b,c,j] / (Sx_j * Sy_j)
// Kernel 1 builds U, V (f16, per-axis max-subtracted softmax numerators) and
// Xp = x / (Sx*Sy) (f16) in workspace.
// Kernel 2: per (b,gy) a 32x64x4096 GEMM on v_wmma_f32_16x16x32_f16 with
// double-buffered LDS tiles; B tile staged via async global->LDS (ASYNCcnt),
// A tile staged via packed-f16 VALU multiply.

typedef __attribute__((ext_vector_type(16))) _Float16 v16h;
typedef __attribute__((ext_vector_type(8)))  _Float16 v8h;
typedef __attribute__((ext_vector_type(8)))  float    v8f;
typedef __attribute__((ext_vector_type(4)))  int      v4i;

typedef __attribute__((address_space(1))) v4i gv4i;   // global int4
typedef __attribute__((address_space(3))) v4i lv4i;   // LDS int4

#define NB   8
#define NC   32
#define HW   4096          // 64*64
#define G    64
#define KT   256           // K tile staged in LDS
#define NROUND (HW / KT)   // 16
#define INVT 10.0f         // 1 / 0.1

// ---------------------------------------------------------------- async utils
__device__ __forceinline__ void wait_async0() {
#if __has_builtin(__builtin_amdgcn_s_wait_asynccnt)
    __builtin_amdgcn_s_wait_asynccnt(0);
#else
    asm volatile("s_wait_asynccnt 0x0" ::: "memory");
#endif
}

// 16-byte global -> LDS copy. Uses the CDNA5 async-DMA path when the builtin
// exists; otherwise a plain vector copy (ds store, DScnt-tracked).
__device__ __forceinline__ void async_copy_b128(const _Float16* g, _Float16* l) {
#if __has_builtin(__builtin_amdgcn_global_load_async_to_lds_b128)
    __builtin_amdgcn_global_load_async_to_lds_b128(
        (gv4i*)(uintptr_t)g,
        (lv4i*)(uint32_t)(uintptr_t)l,
        0, 0);
#else
    *(v8h*)l = *(const v8h*)g;
#endif
}

// ---------------------------------------------------------------- prep kernel
__global__ void __launch_bounds__(256)
swarp_prep_kernel(const float* __restrict__ x, const float* __restrict__ of,
                  _Float16* __restrict__ U, _Float16* __restrict__ V,
                  _Float16* __restrict__ Xp) {
    int t = blockIdx.x * blockDim.x + threadIdx.x;   // 0 .. NB*HW
    if (t >= NB * HW) return;
    int b = t >> 12;
    int j = t & (HW - 1);
    int jx = j >> 6, jy = j & 63;

    float ox = (float)jx + of[(b * 2 + 0) * HW + j];
    float oy = (float)jy + of[(b * 2 + 1) * HW + j];

    // distance to nearest in-range grid point = per-axis softmax max-shift
    float rx = fminf(fmaxf(rintf(ox), 0.0f), 63.0f);
    float ry = fminf(fmaxf(rintf(oy), 0.0f), 63.0f);
    float dmx = fabsf(ox - rx);
    float dmy = fabsf(oy - ry);

    float Sx = 0.0f, Sy = 0.0f;
    #pragma unroll 8
    for (int g = 0; g < G; ++g) {
        float ex = __expf((dmx - fabsf(ox - (float)g)) * INVT);
        float ey = __expf((dmy - fabsf(oy - (float)g)) * INVT);
        Sx += ex;
        Sy += ey;
        U[(b * G + g) * HW + j] = (_Float16)ex;
        V[(b * G + g) * HW + j] = (_Float16)ey;
    }
    float inv = 1.0f / (Sx * Sy);
    #pragma unroll 8
    for (int c = 0; c < NC; ++c) {
        Xp[(b * NC + c) * HW + j] = (_Float16)(x[(b * NC + c) * HW + j] * inv);
    }
}

// ---------------------------------------------------------------- WMMA GEMM
// One block per (gy, b); 8 wave32 waves; wave w owns output tile
// (c-tile = w&1, gx-tile = w>>1).  out[32c x 64gx] = A[32 x 4096] * B[4096 x 64]
// with A[c,k] = Xp[c,k]*V[gy,k], B[k,gx] = U[gx,k].
__global__ void __launch_bounds__(256, 1)
swarp_wmma_kernel(const _Float16* __restrict__ U, const _Float16* __restrict__ V,
                  const _Float16* __restrict__ Xp, float* __restrict__ out) {
    // double-buffered tiles: 2 * (32+64) * 256 halves = 96 KB LDS
    __shared__ __align__(16) _Float16 As[2][NC * KT];  // A[c][k]
    __shared__ __align__(16) _Float16 Bs[2][G * KT];   // B^T[gx][k] = U[gx][k]

    const int gy   = blockIdx.x;
    const int b    = blockIdx.y;
    const int tid  = threadIdx.x;
    const int wave = tid >> 5;
    const int lane = tid & 31;
    const int mt   = wave & 1;   // c-tile index (0..1)
    const int nt   = wave >> 1;  // gx-tile index (0..3)

    const _Float16* __restrict__ Ub   = U  + (size_t)b * G * HW;
    const _Float16* __restrict__ Vrow = V  + ((size_t)b * G + gy) * HW;
    const _Float16* __restrict__ Xb   = Xp + (size_t)b * NC * HW;

    const int arow = lane & 15;   // fragment row / column within 16
    const int ahi  = lane >> 4;   // which K half-group this lane holds

    v8f acc = {};

    // ---- stage one K tile into LDS buffer `buf` (j0 = K start) -------------
    auto stage = [&](int buf, int j0) {
        // A tile: A[c][k] = Xp[c][j0+k] * V[gy][j0+k]; packed f16 multiply.
        // 32*KT/8 = 1024 16B chunks, 4 per thread.
        #pragma unroll
        for (int it = 0; it < (NC * KT) / (8 * 256); ++it) {
            int idx8 = tid + it * 256;
            int c  = idx8 >> 5;          // KT/8 == 32 chunks per row
            int kc = idx8 & 31;
            v8h xv = *(const v8h*)(Xb + c * HW + j0 + kc * 8);
            v8h vv = *(const v8h*)(Vrow + j0 + kc * 8);
            *(v8h*)&As[buf][c * KT + kc * 8] = xv * vv;
        }
        // B tile: pure copy U[gx][j0..j0+KT) -> async global->LDS.
        // 64*KT/8 = 2048 16B chunks, 8 per thread.
        #pragma unroll
        for (int it = 0; it < (G * KT) / (8 * 256); ++it) {
            int idx8 = tid + it * 256;
            int g  = idx8 >> 5;
            int kc = idx8 & 31;
            async_copy_b128(Ub + g * HW + j0 + kc * 8,
                            &Bs[buf][g * KT + kc * 8]);
        }
    };

    // ---- 8 WMMA k-steps over buffer `buf` ----------------------------------
    auto compute = [&](int buf) {
        #pragma unroll
        for (int ks = 0; ks < KT; ks += 32) {
            v16h afrag, bfrag;
            // A 16x32 f16 fragment: lane holds row (mt*16+arow),
            // K chunks {ahi*8..+7, 16+ahi*8..+7}
            const _Float16* ap = &As[buf][(mt * 16 + arow) * KT + ks];
            ((v8h*)&afrag)[0] = *(const v8h*)(ap + ahi * 8);
            ((v8h*)&afrag)[1] = *(const v8h*)(ap + 16 + ahi * 8);
            // B 32x16 f16 fragment: lane holds column (nt*16+arow),
            // K chunks {ahi*8..+7, 16+ahi*8..+7}
            const _Float16* bp = &Bs[buf][(nt * 16 + arow) * KT + ks];
            ((v8h*)&bfrag)[0] = *(const v8h*)(bp + ahi * 8);
            ((v8h*)&bfrag)[1] = *(const v8h*)(bp + 16 + ahi * 8);

            acc = __builtin_amdgcn_wmma_f32_16x16x32_f16(
                /*neg_a=*/false, afrag, /*neg_b=*/false, bfrag,
                /*c_mod=*/(short)0, acc, /*reuse_a=*/false, /*reuse_b=*/false);
        }
    };

    // ---- software pipeline: stage(r+1) overlaps compute(r) -----------------
    stage(0, 0);
    for (int r = 0; r < NROUND; ++r) {
        wait_async0();        // my wave's async loads into buf[r&1] complete
        __syncthreads();      // all waves' staging of buf[r&1] published
        if (r + 1 < NROUND)
            stage((r + 1) & 1, (r + 1) * KT);
        compute(r & 1);
    }

    // D layout: lane holds column N = lane&15, rows M = (lane>>4)*8 + r
    const int n  = lane & 15;
    const int hi = lane >> 4;
    const int gx = nt * 16 + n;
    #pragma unroll
    for (int r = 0; r < 8; ++r) {
        int c = mt * 16 + hi * 8 + r;
        out[(((size_t)b * NC + c) * G + gx) * G + gy] = acc[r];
    }
}

// ---------------------------------------------------------------- launch
extern "C" void kernel_launch(void* const* d_in, const int* in_sizes, int n_in,
                              void* d_out, int out_size, void* d_ws, size_t ws_size,
                              hipStream_t stream) {
    const float* x  = (const float*)d_in[0];   // [8,32,64,64] f32
    const float* of = (const float*)d_in[1];   // [8,2,64,64]  f32
    float* out = (float*)d_out;                // [8,32,64,64] f32

    // Workspace layout (f16): U [8,64,4096] | V [8,64,4096] | Xp [8,32,4096]
    _Float16* U  = (_Float16*)d_ws;
    _Float16* V  = U + (size_t)NB * G * HW;
    _Float16* Xp = V + (size_t)NB * G * HW;

    swarp_prep_kernel<<<dim3((NB * HW + 255) / 256), dim3(256), 0, stream>>>(
        x, of, U, V, Xp);

    swarp_wmma_kernel<<<dim3(G, NB), dim3(256), 0, stream>>>(U, V, Xp, out);
}
